// SVMParamterEstimator_86157043958401
// MI455X (gfx1250) — matrix operand order repaired and checked by reading, not compiled, not run on hardware
//
#include <hip/hip_runtime.h>
#include <hip/hip_bf16.h>
#include <stdint.h>

// ---------------- problem constants (match reference) ----------------
#define NP 64      // particles
#define NH 256     // hidden
#define NE 64      // input size
#define NB 32      // batch
#define NS 128     // seq len
#define PB (NP*NB) // 2048 rows (particle-major: row = p*NB + b)
#define G4 (4*NH)  // 1024 gate width
#define ALPHA_F 0.1f

typedef __bf16 bf16;
typedef bf16  v16bf __attribute__((ext_vector_type(16)));
typedef bf16  v8bf  __attribute__((ext_vector_type(8)));
typedef float v8f   __attribute__((ext_vector_type(8)));

__device__ __forceinline__ bf16 f2bf(float f) {
    union { float f; uint32_t u; } cv; cv.f = f;
    uint32_t u = cv.u;
    uint32_t r = u + 0x7FFFu + ((u >> 16) & 1u);   // round-to-nearest-even
    union { unsigned short s; bf16 b; } o; o.s = (unsigned short)(r >> 16);
    return o.b;
}
__device__ __forceinline__ unsigned short f2bfu(float f) {
    union { float f; uint32_t u; } cv; cv.f = f;
    uint32_t u = cv.u;
    return (unsigned short)((u + 0x7FFFu + ((u >> 16) & 1u)) >> 16);
}
__device__ __forceinline__ float sigm(float x) { return 1.0f / (1.0f + __expf(-x)); }
__device__ __forceinline__ float lrelu(float x) { return x > 0.0f ? x : 0.01f * x; }
__device__ __forceinline__ uint32_t wang(uint32_t x) {
    x = (x ^ 61u) ^ (x >> 16); x *= 9u; x ^= x >> 4; x *= 0x27d4eb2du; x ^= x >> 15; return x;
}

// ---------------- prep: bf16 conversions + obs permute + bias fold ----------------
__global__ void prep_kernel(const float* __restrict__ W_hh, const float* __restrict__ W_ih,
                            const float* __restrict__ obs,  const float* __restrict__ b_ih,
                            const float* __restrict__ b_hh,
                            bf16* __restrict__ Whh_b, bf16* __restrict__ Wih_b,
                            bf16* __restrict__ Xb, float* __restrict__ bias_sum) {
    const int nWhh = G4 * NH;            // 262144
    const int nWih = G4 * NE;            // 65536
    const int nXb  = NS * NB * NE;       // 262144
    const int total = nWhh + nWih + nXb + G4;
    for (int i = blockIdx.x * blockDim.x + threadIdx.x; i < total; i += gridDim.x * blockDim.x) {
        if (i < nWhh) {
            Whh_b[i] = f2bf(W_hh[i]);
        } else if (i < nWhh + nWih) {
            int j = i - nWhh; Wih_b[j] = f2bf(W_ih[j]);
        } else if (i < nWhh + nWih + nXb) {
            int j = i - nWhh - nWih;           // dest index: (t*NB + b)*NE + e
            int e = j & (NE - 1);
            int tb = j / NE; int b = tb & (NB - 1); int t = tb / NB;
            Xb[j] = f2bf(obs[((size_t)b * NS + t) * NE + e]);
        } else {
            int n = i - nWhh - nWih - nXb;
            bias_sum[n] = b_ih[n] + b_hh[n];
        }
    }
}

// ---------------- bf16 WMMA GEMM (one-shot): C[M,N] = A[M,K] * B[N,K]^T + bias[N] ----------------
// Fragment layouts: A elem e -> K = e + (lane>=16?16:0) (one contiguous 16-chunk),
// B elem e -> K = e + (lane>=16?16:0) (one contiguous 16-chunk), so
// plain row-major bf16 feeds both fragments with plain b128 vector loads.
__global__ void wmma_gemm_bias_kernel(const bf16* __restrict__ A, const bf16* __restrict__ Bw,
                                      float* __restrict__ C, int M, int N, int K,
                                      const float* __restrict__ bias) {
    const int lane = threadIdx.x & 31;
    const int wave = threadIdx.x >> 5;
    const int mt = blockIdx.x;                  // 16-row tile
    const int nt = blockIdx.y * 8 + wave;       // 16-col tile
    const int l15  = lane & 15;
    const int hsel = lane >> 4;
    const int rowA = mt * 16 + l15;
    const int rowB = nt * 16 + l15;             // output column index n
    const bf16* aptr = A  + (size_t)rowA * K + hsel * 16;
    const bf16* bptr = Bw + (size_t)rowB * K + hsel * 16;
    v8f acc = {};
    #pragma unroll 4
    for (int kc = 0; kc < K; kc += 32) {
        v8bf alo = *(const v8bf*)(aptr + kc);
        v8bf ahi = *(const v8bf*)(aptr + kc + 8);
        v16bf a = __builtin_shufflevector(alo, ahi, 0,1,2,3,4,5,6,7,8,9,10,11,12,13,14,15);
        v16bf b = *(const v16bf*)(bptr + kc);
        acc = __builtin_amdgcn_wmma_f32_16x16x32_bf16(false, a, false, b, (short)0, acc,
                                                      false, false);
    }
    const int col   = nt * 16 + l15;
    const int rbase = mt * 16 + hsel * 8;
    const float add = bias[col];
    #pragma unroll
    for (int r = 0; r < 8; ++r) {
        C[(size_t)(rbase + r) * N + col] = acc[r] + add;
    }
}

// ---------------- fused recurrent GEMM + LSTM cell (per step) ----------------
// Each wave owns a 16-row tile (mt) x 16-hidden-col tile (jt) and computes ALL FOUR gate
// tiles (cols j, j+256, j+512, j+768) with one shared A fragment and four accumulators.
// i/f/g/o for each element land in the same lane, so the cell update fuses in-register:
// no gates buffer ever touches memory.
__global__ void lstm_step_kernel(const bf16* __restrict__ hb, const bf16* __restrict__ Whh_b,
                                 const float* __restrict__ Xg, int t,
                                 const float* __restrict__ c_cur,
                                 float* __restrict__ h_pre, float* __restrict__ c_pre) {
    const int lane = threadIdx.x & 31;
    const int wave = threadIdx.x >> 5;
    const int mt = blockIdx.x;                 // 16-row tile of PB
    const int jt = blockIdx.y * 8 + wave;      // 16-col tile of NH (16 tiles total)
    const int l15  = lane & 15;
    const int hsel = lane >> 4;
    const int rowA = mt * 16 + l15;
    const bf16* aptr = hb + (size_t)rowA * NH + hsel * 16;
    const int nbase = jt * 16 + l15;
    const bf16* bp0 = Whh_b + (size_t)(0 * NH + nbase) * NH + hsel * 16;
    const bf16* bp1 = Whh_b + (size_t)(1 * NH + nbase) * NH + hsel * 16;
    const bf16* bp2 = Whh_b + (size_t)(2 * NH + nbase) * NH + hsel * 16;
    const bf16* bp3 = Whh_b + (size_t)(3 * NH + nbase) * NH + hsel * 16;
    v8f ai = {}, af = {}, ag = {}, ao = {};
    #pragma unroll 2
    for (int kc = 0; kc < NH; kc += 32) {
        v8bf alo = *(const v8bf*)(aptr + kc);
        v8bf ahi = *(const v8bf*)(aptr + kc + 8);
        v16bf a = __builtin_shufflevector(alo, ahi, 0,1,2,3,4,5,6,7,8,9,10,11,12,13,14,15);
        v16bf b0 = *(const v16bf*)(bp0 + kc);
        v16bf b1 = *(const v16bf*)(bp1 + kc);
        v16bf b2 = *(const v16bf*)(bp2 + kc);
        v16bf b3 = *(const v16bf*)(bp3 + kc);
        ai = __builtin_amdgcn_wmma_f32_16x16x32_bf16(false, a, false, b0, (short)0, ai, false, false);
        af = __builtin_amdgcn_wmma_f32_16x16x32_bf16(false, a, false, b1, (short)0, af, false, false);
        ag = __builtin_amdgcn_wmma_f32_16x16x32_bf16(false, a, false, b2, (short)0, ag, false, false);
        ao = __builtin_amdgcn_wmma_f32_16x16x32_bf16(false, a, false, b3, (short)0, ao, false, false);
    }
    const int j     = jt * 16 + l15;
    const int rbase = mt * 16 + hsel * 8;
    #pragma unroll
    for (int r = 0; r < 8; ++r) {
        const int row = rbase + r;
        const int b   = row & (NB - 1);
        const float* xg = Xg + (size_t)(t * NB + b) * G4;
        const float gi = ai[r] + xg[j];
        const float gf = af[r] + xg[NH + j];
        const float gg = ag[r] + xg[2 * NH + j];
        const float go = ao[r] + xg[3 * NH + j];
        const float c  = c_cur[(size_t)row * NH + j];
        const float c1 = sigm(gf) * c + sigm(gi) * tanhf(gg);
        const float h1 = sigm(go) * tanhf(c1);
        c_pre[(size_t)row * NH + j] = c1;
        h_pre[(size_t)row * NH + j] = h1;
    }
}

// ---------------- per-row dots: logph = h1 . W_obs[E:], sarr = h1 . W_out + b_out ----------------
__global__ void rowdot_kernel(const float* __restrict__ h_pre,
                              const float* __restrict__ W_obs, const float* __restrict__ W_out,
                              const float* __restrict__ b_out,
                              float* __restrict__ logph, float* __restrict__ sarr) {
    __shared__ float r0[NH], r1[NH];
    const int m = blockIdx.x, j = threadIdx.x;
    const float h1 = h_pre[(size_t)m * NH + j];
    r0[j] = h1 * W_obs[NE + j];
    r1[j] = h1 * W_out[j];
    __syncthreads();
    for (int s = NH / 2; s > 0; s >>= 1) {
        if (j < s) { r0[j] += r0[j + s]; r1[j] += r1[j + s]; }
        __syncthreads();
    }
    if (j == 0) { logph[m] = r0[0]; sarr[m] = r1[0] + b_out[0]; }
}

// ---------------- xobs[t*NB+b] = obs[b,t,:] . W_obs[0:E] + b_obs ----------------
__global__ void xobs_kernel(const float* __restrict__ obs, const float* __restrict__ W_obs,
                            const float* __restrict__ b_obs, float* __restrict__ xobs) {
    __shared__ float red[NE];
    const int tb = blockIdx.x;                 // t*NB + b
    const int b = tb & (NB - 1), t = tb / NB;
    const int e = threadIdx.x;
    red[e] = obs[((size_t)b * NS + t) * NE + e] * W_obs[e];
    __syncthreads();
    for (int s = NE / 2; s > 0; s >>= 1) {
        if (e < s) red[e] += red[e + s];
        __syncthreads();
    }
    if (e == 0) xobs[tb] = red[0] + b_obs[0];
}

// ---------------- init state ----------------
__global__ void init_kernel(const float* __restrict__ h0, const float* __restrict__ c0,
                            float* __restrict__ c_cur, bf16* __restrict__ hb,
                            float* __restrict__ p_cur) {
    const int m = blockIdx.x, j = threadIdx.x;
    const size_t idx = (size_t)m * NH + j;
    hb[idx] = f2bf(h0[idx]);
    c_cur[idx] = c0[idx];
    if (j == 0) p_cur[m] = __logf(1.0f / NP);
}

// ---------------- per-batch log-softmax + soft resampling + gather ----------------
__global__ void resample_kernel(int t, const float* __restrict__ h_pre, const float* __restrict__ c_pre,
                                const float* __restrict__ logph, const float* __restrict__ xobs,
                                const float* __restrict__ sarr,
                                float* __restrict__ c_cur, bf16* __restrict__ hb,
                                float* __restrict__ p_cur,
                                float* __restrict__ out_y, float* __restrict__ out_pf) {
    __shared__ float lp[NP], w[NP], cum[NP], wn[NP], pn[NP];
    __shared__ int   src[NP];
    const int b = blockIdx.x, tid = threadIdx.x;
    if (tid < NP) lp[tid] = logph[tid * NB + b] + xobs[t * NB + b] + p_cur[tid * NB + b];
    __syncthreads();
    if (tid == 0) {
        // log-softmax over particle axis
        float mx = lp[0];
        for (int p = 1; p < NP; ++p) mx = fmaxf(mx, lp[p]);
        float se = 0.0f;
        for (int p = 0; p < NP; ++p) se += __expf(lp[p] - mx);
        float lse = mx + __logf(se);
        float csum = 0.0f;
        for (int p = 0; p < NP; ++p) {
            lp[p] -= lse;
            w[p] = ALPHA_F * __expf(lp[p]) + (1.0f - ALPHA_F) / NP;
            csum += w[p];
            cum[p] = csum;
        }
        // deterministic categorical draws (hash RNG; bitwise JAX match impossible)
        for (int p = 0; p < NP; ++p) {
            uint32_t h = wang(0x9e3779b9u ^ ((uint32_t)t << 16) ^ ((uint32_t)b << 8) ^ (uint32_t)p);
            float u = (float)(h >> 8) * (1.0f / 16777216.0f) * csum;
            int idx = 0;
            while (idx < NP - 1 && cum[idx] <= u) ++idx;
            src[p] = idx;
        }
        float sumw = 0.0f;
        for (int p = 0; p < NP; ++p) {
            wn[p] = ALPHA_F * __expf(lp[src[p]]) + (1.0f - ALPHA_F) / NP;
            sumw += wn[p];
        }
        float yacc = 0.0f;
        for (int p = 0; p < NP; ++p) {
            float ww = wn[p] / sumw;
            pn[p] = __logf(ww);
            yacc += ww * sarr[src[p] * NB + b];   // weights sum to 1 -> y_out = lrelu(sum)
        }
        out_y[t * NB + b] = lrelu(yacc);
    }
    __syncthreads();
    if (tid < NP) {
        p_cur[tid * NB + b] = pn[tid];
        out_pf[(size_t)t * PB + tid * NB + b] = lrelu(sarr[src[tid] * NB + b]);
    }
    __syncthreads();
    // gather h/c state rows (vectorized: 64 particles x 64 float4) + packed bf16 h
    for (int idx = tid; idx < NP * (NH / 4); idx += blockDim.x) {
        const int p = idx >> 6, j4 = idx & ((NH / 4) - 1);
        const size_t drow = (size_t)(p * NB + b) * NH + j4 * 4;
        const size_t srow = (size_t)(src[p] * NB + b) * NH + j4 * 4;
        const float4 hv = *(const float4*)(h_pre + srow);
        const float4 cv = *(const float4*)(c_pre + srow);
        uint2 hpack;
        hpack.x = (uint32_t)f2bfu(hv.x) | ((uint32_t)f2bfu(hv.y) << 16);
        hpack.y = (uint32_t)f2bfu(hv.z) | ((uint32_t)f2bfu(hv.w) << 16);
        *(uint2*)(hb + drow) = hpack;
        *(float4*)(c_cur + drow) = cv;
    }
}

// ---------------- host orchestration ----------------
extern "C" void kernel_launch(void* const* d_in, const int* in_sizes, int n_in,
                              void* d_out, int out_size, void* d_ws, size_t ws_size,
                              hipStream_t stream) {
    (void)in_sizes; (void)n_in; (void)out_size; (void)ws_size;
    const float* obs   = (const float*)d_in[0];
    const float* h0    = (const float*)d_in[1];
    const float* c0    = (const float*)d_in[2];
    const float* W_ih  = (const float*)d_in[3];
    const float* b_ih  = (const float*)d_in[4];
    const float* W_hh  = (const float*)d_in[5];
    const float* b_hh  = (const float*)d_in[6];
    const float* W_obs = (const float*)d_in[7];
    const float* b_obs = (const float*)d_in[8];
    const float* W_out = (const float*)d_in[9];
    const float* b_out = (const float*)d_in[10];

    char* w = (char*)d_ws;
    size_t off = 0;
    auto carve = [&](size_t bytes) { void* p = w + off; off += (bytes + 255) & ~size_t(255); return p; };
    bf16*  Whh_b   = (bf16*)carve((size_t)G4 * NH * 2);
    bf16*  Wih_b   = (bf16*)carve((size_t)G4 * NE * 2);
    bf16*  Xb      = (bf16*)carve((size_t)NS * NB * NE * 2);
    float* bias_sum= (float*)carve((size_t)G4 * 4);
    float* Xg      = (float*)carve((size_t)NS * NB * G4 * 4);   // 16 MB
    float* xobs    = (float*)carve((size_t)NS * NB * 4);
    float* c_cur   = (float*)carve((size_t)PB * NH * 4);
    float* h_pre   = (float*)carve((size_t)PB * NH * 4);
    float* c_pre   = (float*)carve((size_t)PB * NH * 4);
    bf16*  hb      = (bf16*)carve((size_t)PB * NH * 2);
    float* p_cur   = (float*)carve((size_t)PB * 4);
    float* logph   = (float*)carve((size_t)PB * 4);
    float* sarr    = (float*)carve((size_t)PB * 4);

    float* out_y  = (float*)d_out;            // (S, B, 1)
    float* out_pf = (float*)d_out + NS * NB;  // (S, P*B, 1)

    prep_kernel<<<1024, 256, 0, stream>>>(W_hh, W_ih, obs, b_ih, b_hh, Whh_b, Wih_b, Xb, bias_sum);

    // input projection for all (t,b): Xg = Xb(4096x64) * Wih^T(64->1024) + (b_ih+b_hh)
    wmma_gemm_bias_kernel<<<dim3(NS * NB / 16, G4 / 128), 256, 0, stream>>>(
        Xb, Wih_b, Xg, NS * NB, G4, NE, bias_sum);

    xobs_kernel<<<NS * NB, NE, 0, stream>>>(obs, W_obs, b_obs, xobs);
    init_kernel<<<PB, NH, 0, stream>>>(h0, c0, c_cur, hb, p_cur);

    for (int t = 0; t < NS; ++t) {
        // fused: gates = h * W_hh^T + Xg row, LSTM cell in-register, no gates buffer
        lstm_step_kernel<<<dim3(PB / 16, (NH / 16) / 8), 256, 0, stream>>>(
            hb, Whh_b, Xg, t, c_cur, h_pre, c_pre);
        rowdot_kernel<<<PB, NH, 0, stream>>>(h_pre, W_obs, W_out, b_out, logph, sarr);
        resample_kernel<<<NB, 256, 0, stream>>>(
            t, h_pre, c_pre, logph, xobs, sarr, c_cur, hb, p_cur, out_y, out_pf);
    }
}
